// TEDClassifier_75866302316877
// MI455X (gfx1250) — compile-verified, matrix operand order checked by reading, not run on hardware
//
#include <hip/hip_runtime.h>
#include <hip/hip_bf16.h>

typedef __bf16 bf16;
typedef __attribute__((ext_vector_type(16))) __bf16 v16bf;
typedef __attribute__((ext_vector_type(8)))  float  v8f;
typedef unsigned int v4u __attribute__((ext_vector_type(4)));
typedef int v8i __attribute__((ext_vector_type(8)));
typedef int v4i __attribute__((ext_vector_type(4)));

union AccU  { v8f   v; float f[8]; };
union FragU { v16bf v; uint4 u[2]; };

static __device__ __forceinline__ float gelu_exact(float v) {
    return 0.5f * v * (1.0f + erff(v * 0.70710678118654752f));
}

// ---------------------------------------------------------------------------
// TDM 2-D tile load: global (row-major, stride0 elems of 2B) -> LDS contiguous.
// Descriptor layout per CDNA5 ISA 8.3/8.4 (D# groups 0/1; groups 2/3 zero).
// 6-arg builtin form: (g0 v4u, g1 v8i, g2 v4i, g3 v4i, g4 v8i, cpol i32)
// ---------------------------------------------------------------------------
static __device__ __forceinline__ void tdm_load_2d(
    unsigned lds_addr, unsigned long long gaddr,
    unsigned tile0, unsigned tile1, unsigned long long stride0)
{
    v4u g0; v8i g1; v4i z4 = {0, 0, 0, 0}; v8i z8 = {0, 0, 0, 0, 0, 0, 0, 0};
    g0[0] = 1u;                                     // count=1 (valid descriptor)
    g0[1] = lds_addr;                               // LDS byte address
    g0[2] = (unsigned)(gaddr & 0xFFFFFFFFull);      // global_addr[31:0]
    g0[3] = (unsigned)((gaddr >> 32) & 0x01FFFFFFull) | 0x80000000u;  // [56:32] | type=2
    g1[0] = 0x10000;                                // data_size=1 (2 bytes)
    g1[1] = (int)((tile0 & 0xFFFFu) << 16);         // tensor_dim0[15:0] (== tile0)
    g1[2] = (int)(((tile0 >> 16) & 0xFFFFu) | ((tile1 & 0xFFFFu) << 16)); // dim0 hi | dim1 lo
    g1[3] = (int)(((tile1 >> 16) & 0xFFFFu) | ((tile0 & 0xFFFFu) << 16)); // dim1 hi | tile_dim0
    g1[4] = (int)(tile1 & 0xFFFFu);                 // tile_dim1 (tile_dim2 = 0)
    g1[5] = (int)(stride0 & 0xFFFFFFFFull);         // tensor_dim0_stride lo
    g1[6] = (int)((stride0 >> 32) & 0xFFFFull);     // tensor_dim0_stride hi
    g1[7] = 0;                                      // tensor_dim1_stride (unused, 2D)
    __builtin_amdgcn_tensor_load_to_lds(g0, g1, z4, z4, z8, 0);
}

// ---------------------------------------------------------------------------
// bf16 WMMA GEMM: C[Mpad x 256] = A[Mpad x K] @ W[K x 256] (+bias) (+=C)
//  - Mpad % 32 == 0, K % 32 == 0 (caller pads), WT = transposed mirror 256 x K
// block = 256 threads (8 wave32). Tile: 32 rows x 256 cols; per wave 2x2
// 16x16 subtiles -> 4x v_wmma_f32_16x16x32_bf16 per K-step.
// LDS double-buffered and fed by the Tensor Data Mover (wave 0 issues,
// s_wait_tensorcnt + workgroup barrier hand off to compute waves).
// ---------------------------------------------------------------------------
__global__ __launch_bounds__(256) void k_gemm_bf16(
    const bf16* __restrict__ A, int lda,
    const bf16* __restrict__ WT,
    float* __restrict__ C,
    int K, const float* __restrict__ bias, int accumulate)
{
    __shared__ __align__(16) bf16 sA[2][32][32];    //  2 x 2 KB
    __shared__ __align__(16) bf16 sB[2][256][32];   //  2 x 16 KB (N-major)
    const int tid  = threadIdx.x;
    const int wave = tid >> 5;
    const int lane = tid & 31;
    const int half = lane >> 4;     // 0: K 0-7/16-23, 1: K 8-15/24-31
    const int lrow = lane & 15;
    const int row0 = blockIdx.x * 32;
    const int col0 = wave * 32;

    AccU acc[2][2];
#pragma unroll
    for (int mi = 0; mi < 2; ++mi)
#pragma unroll
        for (int ni = 0; ni < 2; ++ni)
#pragma unroll
            for (int j = 0; j < 8; ++j) {
                int r = row0 + mi * 16 + half * 8 + j;
                int c = col0 + ni * 16 + lrow;
                acc[mi][ni].f[j] = accumulate ? C[(size_t)r * 256 + c] : 0.0f;
            }

    const unsigned long long Abase = (unsigned long long)(size_t)A + (size_t)row0 * lda * 2;
    const unsigned long long Bbase = (unsigned long long)(size_t)WT;
    const bool issuer = (wave == 0);

    auto stage = [&](int buf, int kb) {
        tdm_load_2d((unsigned)(size_t)&sA[buf][0][0], Abase + (size_t)kb * 2,
                    32u, 32u, (unsigned long long)lda);
        tdm_load_2d((unsigned)(size_t)&sB[buf][0][0], Bbase + (size_t)kb * 2,
                    32u, 256u, (unsigned long long)K);
    };

    if (issuer) stage(0, 0);

    int cur = 0;
    for (int kb = 0; kb < K; kb += 32) {
        __syncthreads();   // buffers[cur^1] no longer being read by compute waves
        if (issuer) {
            if (kb + 32 < K) {
                stage(cur ^ 1, kb + 32);                    // prefetch next tile via TDM
                __builtin_amdgcn_s_wait_tensorcnt(2);       // current tile complete
            } else {
                __builtin_amdgcn_s_wait_tensorcnt(0);       // last tile complete
            }
        }
        __syncthreads();   // buf[cur] visible to all waves

        FragU af[2], bf_[2];
#pragma unroll
        for (int mi = 0; mi < 2; ++mi) {
            const bf16* p = &sA[cur][mi * 16 + lrow][half * 8];
            af[mi].u[0] = *(const uint4*)p;
            af[mi].u[1] = *(const uint4*)(p + 16);
        }
#pragma unroll
        for (int ni = 0; ni < 2; ++ni) {
            const bf16* p = &sB[cur][col0 + ni * 16 + lrow][half * 8];
            bf_[ni].u[0] = *(const uint4*)p;
            bf_[ni].u[1] = *(const uint4*)(p + 16);
        }
#pragma unroll
        for (int mi = 0; mi < 2; ++mi)
#pragma unroll
            for (int ni = 0; ni < 2; ++ni)
                acc[mi][ni].v = __builtin_amdgcn_wmma_f32_16x16x32_bf16(
                    false, af[mi].v, false, bf_[ni].v,
                    (short)0, acc[mi][ni].v, false, false);
        cur ^= 1;
    }

#pragma unroll
    for (int mi = 0; mi < 2; ++mi)
#pragma unroll
        for (int ni = 0; ni < 2; ++ni)
#pragma unroll
            for (int j = 0; j < 8; ++j) {
                int r = row0 + mi * 16 + half * 8 + j;
                int c = col0 + ni * 16 + lrow;
                C[(size_t)r * 256 + c] = acc[mi][ni].f[j] + (bias ? bias[c] : 0.0f);
            }
}

// ---------------------------------------------------------------------------
// Row epilogue (width 256): v = x (+pre)(+bias); LN?; GELU?; (+post); -> out(, bf16)
// flags: bit0 = layernorm, bit1 = gelu
// ---------------------------------------------------------------------------
__global__ __launch_bounds__(256) void k_row_epilogue(
    const float* __restrict__ x, const float* __restrict__ pre_add,
    const float* __restrict__ bias, const float* __restrict__ g,
    const float* __restrict__ beta, const float* __restrict__ post_add,
    float* __restrict__ out, bf16* __restrict__ out_bf, int flags)
{
    __shared__ float red[256];
    const int row = blockIdx.x;
    const int t   = threadIdx.x;
    const size_t idx = (size_t)row * 256 + t;
    float v = x[idx];
    if (pre_add) v += pre_add[idx];
    if (bias)    v += bias[t];
    if (flags & 1) {
        red[t] = v; __syncthreads();
        for (int s = 128; s > 0; s >>= 1) { if (t < s) red[t] += red[t + s]; __syncthreads(); }
        float mean = red[0] * (1.0f / 256.0f); __syncthreads();
        float d = v - mean;
        red[t] = d * d; __syncthreads();
        for (int s = 128; s > 0; s >>= 1) { if (t < s) red[t] += red[t + s]; __syncthreads(); }
        float var = red[0] * (1.0f / 256.0f); __syncthreads();
        v = d * rsqrtf(var + 1e-5f) * g[t] + beta[t];
    }
    if (flags & 2) v = gelu_exact(v);
    if (post_add) v += post_add[idx];
    out[idx] = v;
    if (out_bf) out_bf[idx] = (bf16)v;
}

// ---------------------------------------------------------------------------
__global__ void k_f32_to_bf16(const float* __restrict__ s, bf16* __restrict__ d, long n) {
    long i = (long)blockIdx.x * blockDim.x + threadIdx.x;
    long stride = (long)gridDim.x * blockDim.x;
    for (; i < n; i += stride) d[i] = (bf16)s[i];
}

// Transpose-convert weight: W[K x 256] f32 -> WT[256 x Kpad] bf16 (zero pad)
__global__ __launch_bounds__(256) void k_w_to_bf16_t(
    const float* __restrict__ W, bf16* __restrict__ WT, int K, int Kpad)
{
    const int n = blockIdx.x;           // 0..255 output row
    for (int k = threadIdx.x; k < Kpad; k += 256)
        WT[(size_t)n * Kpad + k] = (k < K) ? (bf16)W[(size_t)k * 256 + n] : (bf16)0.0f;
}

// Row-pad convert: src[rows x w] f32 -> dst[rows x 32] bf16 (zero pad)
__global__ __launch_bounds__(32) void k_pad_rows_bf16(
    const float* __restrict__ s, int w, bf16* __restrict__ d)
{
    const int r = blockIdx.x, t = threadIdx.x;
    d[(size_t)r * 32 + t] = (t < w) ? (bf16)s[(size_t)r * w + t] : (bf16)0.0f;
}

// Build x = [bert | nt_emb[nt] | sr_emb[sr]] as bf16, width 832
__global__ __launch_bounds__(256) void k_build_x(
    const float* __restrict__ bert, const float* __restrict__ nt_emb,
    const float* __restrict__ sr_emb, const int* __restrict__ ntype,
    const int* __restrict__ srole, bf16* __restrict__ x)
{
    const int n = blockIdx.x;
    const int nt = ntype[n], sr = srole[n];
    for (int c = threadIdx.x; c < 832; c += 256) {
        float v;
        if (c < 768)      v = bert[(size_t)n * 768 + c];
        else if (c < 800) v = nt_emb[nt * 32 + (c - 768)];
        else              v = sr_emb[sr * 32 + (c - 800)];
        x[(size_t)n * 832 + c] = (bf16)v;
    }
}

// Edge aggregation: agg[(dst*8+et)*256 + c] += h[src*256 + c]; cnt += 1
__global__ __launch_bounds__(256) void k_edge_agg(
    const int* __restrict__ src, const int* __restrict__ dst,
    const int* __restrict__ et, const float* __restrict__ h,
    float* __restrict__ agg, float* __restrict__ cnt, int E)
{
    int e = blockIdx.x * 8 + (threadIdx.x >> 5);
    int lane = threadIdx.x & 31;
    if (e >= E) return;
    int s = src[e], d = dst[e];
    int r = et[e]; r = r < 0 ? 0 : (r > 7 ? 7 : r);
    const float* hs = h + (size_t)s * 256;
    float* ag = agg + ((size_t)d * 8 + r) * 256;
#pragma unroll
    for (int i = 0; i < 8; ++i) atomicAdd(&ag[lane + i * 32], hs[lane + i * 32]);
    if (lane == 0) atomicAdd(&cnt[(size_t)d * 8 + r], 1.0f);
}

__global__ __launch_bounds__(256) void k_agg_norm(
    const float* __restrict__ agg, const float* __restrict__ cnt, bf16* __restrict__ abf)
{
    const size_t i = blockIdx.x;
    const float inv = 1.0f / fmaxf(cnt[i], 1.0f);
    abf[i * 256 + threadIdx.x] = (bf16)(agg[i * 256 + threadIdx.x] * inv);
}

// Pools: order [claim, td, cs, evid, pro, opp, graph]
__global__ __launch_bounds__(256) void k_pool_accum(
    const float* __restrict__ nr, const int* __restrict__ ntype,
    const int* __restrict__ srole, const int* __restrict__ batch,
    float* __restrict__ sums, float* __restrict__ cnts)
{
    const int n = blockIdx.x, t = threadIdx.x;
    const int nt = ntype[n], sr = srole[n], b = batch[n];
    bool pro = (sr == 1) || (sr == 3) || (sr == 5);
    bool opp = (sr == 2) || (sr == 4) || (sr == 6);
    bool m[7] = { nt == 0, nt == 1, nt == 2, nt == 5,
                  (nt == 3) && pro, (nt == 3) && opp, true };
    float v = nr[(size_t)n * 256 + t];
#pragma unroll
    for (int j = 0; j < 7; ++j)
        if (m[j]) atomicAdd(&sums[((size_t)b * 7 + j) * 256 + t], v);
    if (t == 0)
#pragma unroll
        for (int j = 0; j < 7; ++j)
            if (m[j]) atomicAdd(&cnts[b * 7 + j], 1.0f);
}

__global__ __launch_bounds__(256) void k_pool_fin(
    const float* __restrict__ sums, const float* __restrict__ cnts, float* __restrict__ pooled)
{
    const size_t i = blockIdx.x;
    pooled[i * 256 + threadIdx.x] = sums[i * 256 + threadIdx.x] / fmaxf(cnts[i], 1.0f);
}

__global__ void k_gather_src(const float* __restrict__ emb, const int* __restrict__ ids,
                             bf16* __restrict__ d) {
    d[(size_t)blockIdx.x * 128 + threadIdx.x] =
        (bf16)emb[(size_t)ids[blockIdx.x] * 128 + threadIdx.x];
}

// tok order: [claim(p0), evid(p3), pro(p4), opp(p5), td(p1), cs(p2), news_p]
__global__ __launch_bounds__(256) void k_build_tok(
    const float* __restrict__ pooled, const float* __restrict__ news_p,
    float* __restrict__ tokf, bf16* __restrict__ tokbf)
{
    const int b = blockIdx.x, t = threadIdx.x;
    const int map[6] = {0, 3, 4, 5, 1, 2};
    for (int j = 0; j < 7; ++j) {
        float v = (j < 6) ? pooled[((size_t)b * 7 + map[j]) * 256 + t]
                          : news_p[(size_t)b * 256 + t];
        size_t o = ((size_t)b * 7 + j) * 256 + t;
        tokf[o]  = v;
        tokbf[o] = (bf16)v;
    }
}

// 4-head attention over 7 tokens, dh=64. grid=(64,4), block=64
__global__ __launch_bounds__(64) void k_attn(
    const float* __restrict__ Q, const float* __restrict__ K,
    const float* __restrict__ V, float* __restrict__ ctx)
{
    const int b = blockIdx.x, hh = blockIdx.y, t = threadIdx.x;
    __shared__ float sc[7][7];
    __shared__ float red[64];
    float q[7], k[7], v[7];
#pragma unroll
    for (int i = 0; i < 7; ++i) {
        size_t o = ((size_t)(b * 7 + i)) * 256 + hh * 64 + t;
        q[i] = Q[o]; k[i] = K[o]; v[i] = V[o];
    }
    for (int i = 0; i < 7; ++i)
        for (int j = 0; j < 7; ++j) {
            red[t] = q[i] * k[j];
            __syncthreads();
            for (int s = 32; s > 0; s >>= 1) { if (t < s) red[t] += red[t + s]; __syncthreads(); }
            if (t == 0) sc[i][j] = red[0] * 0.125f;   // 1/sqrt(64)
            __syncthreads();
        }
    if (t < 7) {
        float mx = sc[t][0];
        for (int j = 1; j < 7; ++j) mx = fmaxf(mx, sc[t][j]);
        float sum = 0.0f;
        for (int j = 0; j < 7; ++j) { sc[t][j] = expf(sc[t][j] - mx); sum += sc[t][j]; }
        float inv = 1.0f / sum;
        for (int j = 0; j < 7; ++j) sc[t][j] *= inv;
    }
    __syncthreads();
    for (int i = 0; i < 7; ++i) {
        float o = 0.0f;
#pragma unroll
        for (int j = 0; j < 7; ++j) o += sc[i][j] * v[j];
        ctx[((size_t)(b * 7 + i)) * 256 + hh * 64 + t] = o;
    }
}

__global__ __launch_bounds__(256) void k_cat2_bf(
    const float* __restrict__ a, long sa, const float* __restrict__ b, long sb,
    bf16* __restrict__ d)
{
    const int r = blockIdx.x, t = threadIdx.x;
    d[(size_t)r * 512 + t]       = (bf16)a[(size_t)r * sa + t];
    d[(size_t)r * 512 + 256 + t] = (bf16)b[(size_t)r * sb + t];
}

__global__ __launch_bounds__(256) void k_conf_bf(
    const float* __restrict__ pooled, bf16* __restrict__ d)
{
    const int b = blockIdx.x, t = threadIdx.x;
    float p = pooled[((size_t)b * 7 + 4) * 256 + t];
    float o = pooled[((size_t)b * 7 + 5) * 256 + t];
    size_t r = (size_t)b * 1024;
    d[r + t]        = (bf16)p;
    d[r + 256 + t]  = (bf16)o;
    d[r + 512 + t]  = (bf16)(p - o);
    d[r + 768 + t]  = (bf16)(p * o);
}

__global__ __launch_bounds__(256) void k_gin_bf(
    const float* __restrict__ news_p, const float* __restrict__ pooled,
    const float* __restrict__ debate, const float* __restrict__ conflict,
    const float* __restrict__ tok2, const float* __restrict__ meta,
    const float* __restrict__ teach, bf16* __restrict__ d)
{
    const int b = blockIdx.x, t = threadIdx.x;
    size_t r = (size_t)b * 2560;
    d[r + 0 * 256 + t] = (bf16)news_p[(size_t)b * 256 + t];
    d[r + 1 * 256 + t] = (bf16)pooled[((size_t)b * 7 + 0) * 256 + t];  // claim
    d[r + 2 * 256 + t] = (bf16)pooled[((size_t)b * 7 + 1) * 256 + t];  // td
    d[r + 3 * 256 + t] = (bf16)pooled[((size_t)b * 7 + 2) * 256 + t];  // cs
    d[r + 4 * 256 + t] = (bf16)debate[(size_t)b * 256 + t];
    d[r + 5 * 256 + t] = (bf16)conflict[(size_t)b * 256 + t];
    d[r + 6 * 256 + t] = (bf16)tok2[((size_t)b * 7 + 1) * 256 + t];    // evid2
    d[r + 7 * 256 + t] = (bf16)pooled[((size_t)b * 7 + 6) * 256 + t];  // graph
    d[r + 8 * 256 + t] = (bf16)meta[(size_t)b * 256 + t];
    d[r + 9 * 256 + t] = (bf16)teach[(size_t)b * 256 + t];
}

__global__ __launch_bounds__(256) void k_alpha(
    const float* __restrict__ pre, const float* __restrict__ gW2,
    const float* __restrict__ gb2, float* __restrict__ alphas)
{
    __shared__ float red[256];
    const int b = blockIdx.x, g = blockIdx.y, t = threadIdx.x;
    red[t] = pre[((size_t)g * 64 + b) * 256 + t] * gW2[g * 256 + t];
    __syncthreads();
    for (int s = 128; s > 0; s >>= 1) { if (t < s) red[t] += red[t + s]; __syncthreads(); }
    if (t == 0) alphas[b * 6 + g] = 1.0f / (1.0f + expf(-(red[0] + gb2[g])));
}

__global__ __launch_bounds__(256) void k_fused_bf(
    const float* __restrict__ al, const float* __restrict__ news_p,
    const float* __restrict__ debate, const float* __restrict__ pooled,
    const float* __restrict__ tok2, bf16* __restrict__ d)
{
    const int b = blockIdx.x, t = threadIdx.x;
    size_t r = (size_t)b * 1536;
    d[r + 0 * 256 + t] = (bf16)(al[b * 6 + 0] * news_p[(size_t)b * 256 + t]);
    d[r + 1 * 256 + t] = (bf16)(al[b * 6 + 1] * debate[(size_t)b * 256 + t]);
    d[r + 2 * 256 + t] = (bf16)(al[b * 6 + 2] * pooled[((size_t)b * 7 + 1) * 256 + t]);
    d[r + 3 * 256 + t] = (bf16)(al[b * 6 + 3] * pooled[((size_t)b * 7 + 2) * 256 + t]);
    d[r + 4 * 256 + t] = (bf16)(al[b * 6 + 4] * tok2[((size_t)b * 7 + 1) * 256 + t]);
    d[r + 5 * 256 + t] = (bf16)(al[b * 6 + 5] * pooled[((size_t)b * 7 + 6) * 256 + t]);
}

__global__ __launch_bounds__(256) void k_final(
    const float* __restrict__ finh, const float* __restrict__ W2,
    const float* __restrict__ b2, float* __restrict__ out)
{
    __shared__ float red[256];
    const int b = blockIdx.x, t = threadIdx.x;
    float hv = finh[(size_t)b * 256 + t];
    for (int c = 0; c < 2; ++c) {
        red[t] = hv * W2[t * 2 + c];
        __syncthreads();
        for (int s = 128; s > 0; s >>= 1) { if (t < s) red[t] += red[t + s]; __syncthreads(); }
        if (t == 0) out[b * 2 + c] = red[0] + b2[c];
        __syncthreads();
    }
}

// ---------------------------------------------------------------------------
extern "C" void kernel_launch(void* const* d_in, const int* in_sizes, int n_in,
                              void* d_out, int out_size, void* d_ws, size_t ws_size,
                              hipStream_t stream) {
    (void)n_in; (void)out_size; (void)ws_size;
    const float* bert      = (const float*)d_in[0];
    const float* news      = (const float*)d_in[1];
    const float* timef     = (const float*)d_in[2];
    const float* teachf    = (const float*)d_in[3];
    const float* nt_emb    = (const float*)d_in[4];
    const float* sr_emb    = (const float*)d_in[5];
    const float* in_W      = (const float*)d_in[6];
    const float* in_b      = (const float*)d_in[7];
    const float* in_g      = (const float*)d_in[8];
    const float* in_beta   = (const float*)d_in[9];
    const float* rgcn_Wr   = (const float*)d_in[10];
    const float* rgcn_Wroot= (const float*)d_in[11];
    const float* rgcn_b    = (const float*)d_in[12];
    const float* rgcn_g    = (const float*)d_in[13];
    const float* rgcn_beta = (const float*)d_in[14];
    const float* np_W      = (const float*)d_in[15];
    const float* np_b      = (const float*)d_in[16];
    const float* news_W    = (const float*)d_in[17];
    const float* news_b    = (const float*)d_in[18];
    const float* news_g    = (const float*)d_in[19];
    const float* news_be   = (const float*)d_in[20];
    const float* src_emb   = (const float*)d_in[21];
    const float* src_W     = (const float*)d_in[22];
    const float* src_b     = (const float*)d_in[23];
    const float* src_g     = (const float*)d_in[24];
    const float* src_be    = (const float*)d_in[25];
    const float* time_W    = (const float*)d_in[26];
    const float* time_b    = (const float*)d_in[27];
    const float* time_g    = (const float*)d_in[28];
    const float* time_be   = (const float*)d_in[29];
    const float* meta_W    = (const float*)d_in[30];
    const float* meta_b    = (const float*)d_in[31];
    const float* meta_g    = (const float*)d_in[32];
    const float* meta_be   = (const float*)d_in[33];
    const float* teach_W   = (const float*)d_in[34];
    const float* teach_b   = (const float*)d_in[35];
    const float* teach_g   = (const float*)d_in[36];
    const float* teach_be  = (const float*)d_in[37];
    const float* Wq        = (const float*)d_in[38];
    const float* Wk        = (const float*)d_in[39];
    const float* Wv        = (const float*)d_in[40];
    const float* Wo        = (const float*)d_in[41];
    const float* bq        = (const float*)d_in[42];
    const float* bk        = (const float*)d_in[43];
    const float* bv        = (const float*)d_in[44];
    const float* bo        = (const float*)d_in[45];
    const float* attn_g    = (const float*)d_in[46];
    const float* attn_be   = (const float*)d_in[47];
    const float* gW1       = (const float*)d_in[48];
    const float* gb1       = (const float*)d_in[49];
    const float* gW2       = (const float*)d_in[50];
    const float* gb2       = (const float*)d_in[51];
    const float* deb_W     = (const float*)d_in[52];
    const float* deb_b     = (const float*)d_in[53];
    const float* deb_g     = (const float*)d_in[54];
    const float* deb_be    = (const float*)d_in[55];
    const float* conf_W    = (const float*)d_in[56];
    const float* conf_b    = (const float*)d_in[57];
    const float* conf_g    = (const float*)d_in[58];
    const float* conf_be   = (const float*)d_in[59];
    const float* fin_W1    = (const float*)d_in[60];
    const float* fin_b1    = (const float*)d_in[61];
    const float* fin_g     = (const float*)d_in[62];
    const float* fin_be    = (const float*)d_in[63];
    const float* fin_W2    = (const float*)d_in[64];
    const float* fin_b2    = (const float*)d_in[65];
    const int* node_type   = (const int*)d_in[66];
    const int* speaker_role= (const int*)d_in[67];
    const int* edge_index  = (const int*)d_in[68];
    const int* edge_type   = (const int*)d_in[69];
    const int* batch       = (const int*)d_in[70];
    const int* source_ids  = (const int*)d_in[71];

    const int N   = in_sizes[66];
    const int E   = in_sizes[69];
    const int N32 = (N + 31) & ~31;

    char* base = (char*)d_ws;
    size_t off = 0;
    auto alloc = [&](size_t bytes) -> void* {
        void* p = base + off;
        off += (bytes + 255) & ~(size_t)255;
        return p;
    };

    // transposed bf16 weight mirrors (256 x Kpad each)
    bf16* in_WT   = (bf16*)alloc((size_t)256 * 832 * 2);
    bf16* WrT     = (bf16*)alloc((size_t)2 * 256 * 2048 * 2);
    bf16* WrootT  = (bf16*)alloc((size_t)2 * 256 * 256 * 2);
    bf16* npWT    = (bf16*)alloc((size_t)256 * 256 * 2);
    bf16* newsWT  = (bf16*)alloc((size_t)256 * 768 * 2);
    bf16* srcWT   = (bf16*)alloc((size_t)256 * 128 * 2);
    bf16* timeWT  = (bf16*)alloc((size_t)256 * 32 * 2);
    bf16* metaWT  = (bf16*)alloc((size_t)256 * 512 * 2);
    bf16* teachWT = (bf16*)alloc((size_t)256 * 32 * 2);
    bf16* WqT     = (bf16*)alloc((size_t)256 * 256 * 2);
    bf16* WkT     = (bf16*)alloc((size_t)256 * 256 * 2);
    bf16* WvT     = (bf16*)alloc((size_t)256 * 256 * 2);
    bf16* WoT     = (bf16*)alloc((size_t)256 * 256 * 2);
    bf16* gW1T    = (bf16*)alloc((size_t)6 * 256 * 2560 * 2);
    bf16* debWT   = (bf16*)alloc((size_t)256 * 512 * 2);
    bf16* confWT  = (bf16*)alloc((size_t)256 * 1024 * 2);
    bf16* finW1T  = (bf16*)alloc((size_t)256 * 1536 * 2);

    // large activations (node dim padded to N32 rows)
    float* h      = (float*)alloc((size_t)N32 * 256 * 4);
    bf16*  h_bf   = (bf16*) alloc((size_t)N32 * 256 * 2);
    float* m_buf  = (float*)alloc((size_t)N32 * 256 * 4);
    float* nrp    = (float*)alloc((size_t)N32 * 256 * 4);
    float* agg    = (float*)alloc((size_t)N32 * 8 * 256 * 4);
    bf16*  agg_bf = (bf16*) alloc((size_t)N32 * 8 * 256 * 2);
    float* cnt    = (float*)alloc((size_t)N * 8 * 4);
    bf16*  x_cat  = (bf16*)agg;   // alias: x only live before agg is zeroed

    // small buffers
    float* sums    = (float*)alloc((size_t)64 * 7 * 256 * 4);
    float* cnts    = (float*)alloc((size_t)64 * 7 * 4);
    float* pooled  = (float*)alloc((size_t)64 * 7 * 256 * 4);
    float* tokf    = (float*)alloc((size_t)448 * 256 * 4);
    bf16*  tokbf   = (bf16*) alloc((size_t)448 * 256 * 2);
    float* Qb      = (float*)alloc((size_t)448 * 256 * 4);
    float* Kb      = (float*)alloc((size_t)448 * 256 * 4);
    float* Vb      = (float*)alloc((size_t)448 * 256 * 4);
    float* ctx     = (float*)alloc((size_t)448 * 256 * 4);
    bf16*  ctxbf   = (bf16*) alloc((size_t)448 * 256 * 2);
    float* tok2    = (float*)alloc((size_t)448 * 256 * 4);
    float* sC      = (float*)alloc((size_t)448 * 256 * 4);
    bf16*  newsb   = (bf16*) alloc((size_t)64 * 768 * 2);
    bf16*  timeb   = (bf16*) alloc((size_t)64 * 32 * 2);
    bf16*  teachb  = (bf16*) alloc((size_t)64 * 32 * 2);
    bf16*  srcb    = (bf16*) alloc((size_t)64 * 128 * 2);
    bf16*  metaib  = (bf16*) alloc((size_t)64 * 512 * 2);
    bf16*  debib   = (bf16*) alloc((size_t)64 * 512 * 2);
    bf16*  confib  = (bf16*) alloc((size_t)64 * 1024 * 2);
    bf16*  ginb    = (bf16*) alloc((size_t)64 * 2560 * 2);
    bf16*  fusedb  = (bf16*) alloc((size_t)64 * 1536 * 2);
    float* news_p  = (float*)alloc((size_t)64 * 256 * 4);
    float* sourcev = (float*)alloc((size_t)64 * 256 * 4);
    float* timev   = (float*)alloc((size_t)64 * 256 * 4);
    float* metav   = (float*)alloc((size_t)64 * 256 * 4);
    float* teachv  = (float*)alloc((size_t)64 * 256 * 4);
    float* debatev = (float*)alloc((size_t)64 * 256 * 4);
    float* conflv  = (float*)alloc((size_t)64 * 256 * 4);
    float* preG    = (float*)alloc((size_t)6 * 64 * 256 * 4);
    float* alphas  = (float*)alloc((size_t)64 * 6 * 4);
    float* finh    = (float*)alloc((size_t)64 * 256 * 4);

    auto cvt = [&](const float* s, bf16* d, long n) {
        long g = (n + 255) / 256; if (g > 65535) g = 65535;
        k_f32_to_bf16<<<dim3((unsigned)g), 256, 0, stream>>>(s, d, n);
    };
    auto wtr = [&](const float* W, bf16* WT, int K, int Kpad) {
        k_w_to_bf16_t<<<dim3(256), 256, 0, stream>>>(W, WT, K, Kpad);
    };
    auto gemm = [&](const bf16* A, int lda, const bf16* WT, float* C, int Mpad, int K,
                    const float* bias, int accum) {
        k_gemm_bf16<<<dim3(Mpad / 32), 256, 0, stream>>>(A, lda, WT, C, K, bias, accum);
    };
    auto epi = [&](const float* x, const float* pre, const float* bias, const float* g,
                   const float* be, const float* post, float* out, bf16* obf, int M, int flags) {
        k_row_epilogue<<<dim3(M), 256, 0, stream>>>(x, pre, bias, g, be, post, out, obf, flags);
    };

    // weight transposes / conversions
    wtr(in_W, in_WT, 832, 832);
    for (int l = 0; l < 2; ++l) {
        wtr(rgcn_Wr + (size_t)l * 2048 * 256, WrT + (size_t)l * 256 * 2048, 2048, 2048);
        wtr(rgcn_Wroot + (size_t)l * 256 * 256, WrootT + (size_t)l * 256 * 256, 256, 256);
    }
    wtr(np_W, npWT, 256, 256);
    wtr(news_W, newsWT, 768, 768);
    wtr(src_W, srcWT, 128, 128);
    wtr(time_W, timeWT, 8, 32);
    wtr(meta_W, metaWT, 512, 512);
    wtr(teach_W, teachWT, 4, 32);
    wtr(Wq, WqT, 256, 256); wtr(Wk, WkT, 256, 256);
    wtr(Wv, WvT, 256, 256); wtr(Wo, WoT, 256, 256);
    for (int gi = 0; gi < 6; ++gi)
        wtr(gW1 + (size_t)gi * 2560 * 256, gW1T + (size_t)gi * 256 * 2560, 2560, 2560);
    wtr(deb_W, debWT, 512, 512);
    wtr(conf_W, confWT, 1024, 1024);
    wtr(fin_W1, finW1T, 1536, 1536);

    // input projection: h = gelu(ln(x @ in_W + b))
    k_build_x<<<dim3(N), 256, 0, stream>>>(bert, nt_emb, sr_emb, node_type, speaker_role, x_cat);
    gemm(x_cat, 832, in_WT, m_buf, N32, 832, nullptr, 0);
    epi(m_buf, nullptr, in_b, in_g, in_beta, nullptr, h, h_bf, N, 3);

    // RGCN layers: relation-stacked GEMM K=2048 (+root K=256 accumulated)
    for (int l = 0; l < 2; ++l) {
        (void)hipMemsetAsync(agg, 0, (size_t)N * 8 * 256 * 4, stream);
        (void)hipMemsetAsync(cnt, 0, (size_t)N * 8 * 4, stream);
        k_edge_agg<<<dim3((E + 7) / 8), 256, 0, stream>>>(
            edge_index, edge_index + E, edge_type, h, agg, cnt, E);
        k_agg_norm<<<dim3(N * 8), 256, 0, stream>>>(agg, cnt, agg_bf);
        gemm(agg_bf, 2048, WrT + (size_t)l * 256 * 2048, m_buf, N32, 2048, nullptr, 0);
        gemm(h_bf, 256, WrootT + (size_t)l * 256 * 256, m_buf, N32, 256, nullptr, 1);
        epi(m_buf, nullptr, rgcn_b + l * 256, rgcn_g + l * 256, rgcn_beta + l * 256,
            h, h, h_bf, N, 3);
    }

    // node projection + pools
    gemm(h_bf, 256, npWT, nrp, N32, 256, np_b, 0);
    (void)hipMemsetAsync(sums, 0, (size_t)64 * 7 * 256 * 4, stream);
    (void)hipMemsetAsync(cnts, 0, (size_t)64 * 7 * 4, stream);
    k_pool_accum<<<dim3(N), 256, 0, stream>>>(nrp, node_type, speaker_role, batch, sums, cnts);
    k_pool_fin<<<dim3(64 * 7), 256, 0, stream>>>(sums, cnts, pooled);

    // feature MLPs
    cvt(news, newsb, 64L * 768);
    gemm(newsb, 768, newsWT, sC, 64, 768, nullptr, 0);
    epi(sC, nullptr, news_b, news_g, news_be, nullptr, news_p, nullptr, 64, 3);

    k_gather_src<<<dim3(64), 128, 0, stream>>>(src_emb, source_ids, srcb);
    gemm(srcb, 128, srcWT, sC, 64, 128, nullptr, 0);
    epi(sC, nullptr, src_b, src_g, src_be, nullptr, sourcev, nullptr, 64, 3);

    k_pad_rows_bf16<<<dim3(64), 32, 0, stream>>>(timef, 8, timeb);
    gemm(timeb, 32, timeWT, sC, 64, 32, nullptr, 0);
    epi(sC, nullptr, time_b, time_g, time_be, nullptr, timev, nullptr, 64, 3);

    k_cat2_bf<<<dim3(64), 256, 0, stream>>>(sourcev, 256, timev, 256, metaib);
    gemm(metaib, 512, metaWT, sC, 64, 512, nullptr, 0);
    epi(sC, nullptr, meta_b, meta_g, meta_be, nullptr, metav, nullptr, 64, 3);

    k_pad_rows_bf16<<<dim3(64), 32, 0, stream>>>(teachf, 4, teachb);
    gemm(teachb, 32, teachWT, sC, 64, 32, nullptr, 0);
    epi(sC, nullptr, teach_b, teach_g, teach_be, nullptr, teachv, nullptr, 64, 3);

    // attention over 7 tokens
    k_build_tok<<<dim3(64), 256, 0, stream>>>(pooled, news_p, tokf, tokbf);
    gemm(tokbf, 256, WqT, Qb, 448, 256, bq, 0);
    gemm(tokbf, 256, WkT, Kb, 448, 256, bk, 0);
    gemm(tokbf, 256, WvT, Vb, 448, 256, bv, 0);
    k_attn<<<dim3(64, 4), 64, 0, stream>>>(Qb, Kb, Vb, ctx);
    cvt(ctx, ctxbf, 448L * 256);
    gemm(ctxbf, 256, WoT, sC, 448, 256, bo, 0);
    epi(sC, tokf, nullptr, attn_g, attn_be, nullptr, tok2, nullptr, 448, 1);

    // debate & conflict MLPs
    k_cat2_bf<<<dim3(64), 256, 0, stream>>>(pooled + 1 * 256, 7 * 256,
                                            pooled + 2 * 256, 7 * 256, debib);
    gemm(debib, 512, debWT, sC, 64, 512, nullptr, 0);
    epi(sC, nullptr, deb_b, deb_g, deb_be, nullptr, debatev, nullptr, 64, 3);

    k_conf_bf<<<dim3(64), 256, 0, stream>>>(pooled, confib);
    gemm(confib, 1024, confWT, sC, 64, 1024, nullptr, 0);
    epi(sC, nullptr, conf_b, conf_g, conf_be, nullptr, conflv, nullptr, 64, 3);

    // gates
    k_gin_bf<<<dim3(64), 256, 0, stream>>>(news_p, pooled, debatev, conflv, tok2,
                                           metav, teachv, ginb);
    for (int gi = 0; gi < 6; ++gi) {
        float* pg = preG + (size_t)gi * 64 * 256;
        gemm(ginb, 2560, gW1T + (size_t)gi * 256 * 2560, pg, 64, 2560, nullptr, 0);
        epi(pg, nullptr, gb1 + gi * 256, nullptr, nullptr, nullptr, pg, nullptr, 64, 2);
    }
    k_alpha<<<dim3(64, 6), 256, 0, stream>>>(preG, gW2, gb2, alphas);

    // fusion + head
    k_fused_bf<<<dim3(64), 256, 0, stream>>>(alphas, news_p, debatev, pooled, tok2, fusedb);
    gemm(fusedb, 1536, finW1T, sC, 64, 1536, nullptr, 0);
    epi(sC, nullptr, fin_b1, fin_g, fin_be, nullptr, finh, nullptr, 64, 3);
    k_final<<<dim3(64), 256, 0, stream>>>(finh, fin_W2, fin_b2, (float*)d_out);
}
